// GPT_69020124446834
// MI455X (gfx1250) — matrix operand order repaired and checked by reading, not compiled, not run on hardware
//
#include <hip/hip_runtime.h>
#include <cstdint>
#include <cstddef>

// ---------------------------------------------------------------------------
// GPT forward for MI455X (gfx1250, wave32, WMMA).
// All GEMMs run on v_wmma_f32_16x16x32_bf16 (fp32 params converted to bf16,
// f32 accumulation). GEMM global->LDS staging uses CDNA5 async-to-LDS copies
// (ASYNCcnt) with double buffering. Attention is flash-style, one wave per
// 16-row q tile.
// ---------------------------------------------------------------------------

typedef __attribute__((ext_vector_type(16))) __bf16 v16bf;
typedef __attribute__((ext_vector_type(8)))  __bf16 bf16x8;
typedef __attribute__((ext_vector_type(8)))  float  v8f;

#define DEVI __device__ __forceinline__

constexpr int B_  = 4;
constexpr int T_  = 1024;
constexpr int E_  = 512;
constexpr int D_  = 1024;
constexpr int NH_ = 16;
constexpr int DH_ = 64;      // D/NH
constexpr int FF_ = 4096;
constexpr int AD_ = 8;
constexpr int NL_ = 8;
constexpr int M_  = B_ * T_; // 4096 rows
constexpr float EPS_ = 1e-5f;

// Build a v16bf fragment from two 16-byte (8 x bf16) chunks.
DEVI v16bf frag_from2(const __bf16* p0, const __bf16* p1) {
  v16bf r;
  *(bf16x8*)&r       = *(const bf16x8*)p0;
  *((bf16x8*)&r + 1) = *(const bf16x8*)p1;
  return r;
}

// Async global -> LDS copy of 32 contiguous bytes (two B128 beats), per lane.
// The instruction offset applies to both the LDS and the global address.
DEVI void async_copy32(uint32_t lds_off, const __bf16* gptr) {
  asm volatile(
      "global_load_async_to_lds_b128 %0, %1, off\n\t"
      "global_load_async_to_lds_b128 %0, %1, off offset:16"
      :: "v"(lds_off), "v"(gptr)
      : "memory");
}

// ---------------------------------------------------------------------------
// Embedding: h[row, 0:512]   = shifted_lcd(row) @ W_embed
//            h[row, 512:1024]= action(row) @ W_act     ; += pos_emb
// ---------------------------------------------------------------------------
__global__ __launch_bounds__(256) void k_embed(
    const float* __restrict__ lcd, const float* __restrict__ act,
    const float* __restrict__ pos, const float* __restrict__ Wemb,
    const float* __restrict__ Wact, float* __restrict__ h)
{
  const int row = blockIdx.x;          // b*T + t
  const int b = row >> 10, t = row & (T_ - 1);
  const float* lrow = (t > 0) ? (lcd + (size_t)(b * T_ + t - 1) * E_) : nullptr;
  const float* arow = act + (size_t)row * AD_;
  for (int c = threadIdx.x; c < D_ / 2; c += blockDim.x) {
    float s = 0.f;
    if (lrow) {
      #pragma unroll 4
      for (int e = 0; e < E_; ++e) s += lrow[e] * Wemb[(size_t)e * (D_ / 2) + c];
    }
    h[(size_t)row * D_ + c] = s + pos[(size_t)t * D_ + c];
    float s2 = 0.f;
    #pragma unroll
    for (int a = 0; a < AD_; ++a) s2 += arow[a] * Wact[a * (D_ / 2) + c];
    h[(size_t)row * D_ + (D_ / 2) + c] = s2 + pos[(size_t)t * D_ + (D_ / 2) + c];
  }
}

// ---------------------------------------------------------------------------
// Row LayerNorm over D=1024, fp32 in -> bf16 out (GEMM operand).
// ---------------------------------------------------------------------------
__global__ __launch_bounds__(256) void k_ln(
    const float* __restrict__ x, const float* __restrict__ sc,
    const float* __restrict__ bi, __bf16* __restrict__ out)
{
  __shared__ float r1[256], r2[256];
  const int row = blockIdx.x;
  const float* xr = x + (size_t)row * D_;
  float a = 0.f, b = 0.f;
  for (int i = threadIdx.x; i < D_; i += 256) { float v = xr[i]; a += v; b += v * v; }
  r1[threadIdx.x] = a; r2[threadIdx.x] = b; __syncthreads();
  for (int st = 128; st > 0; st >>= 1) {
    if (threadIdx.x < st) {
      r1[threadIdx.x] += r1[threadIdx.x + st];
      r2[threadIdx.x] += r2[threadIdx.x + st];
    }
    __syncthreads();
  }
  const float mean = r1[0] * (1.0f / D_);
  const float var  = r2[0] * (1.0f / D_) - mean * mean;
  const float inv  = rsqrtf(var + EPS_);
  for (int i = threadIdx.x; i < D_; i += 256) {
    float v = (xr[i] - mean) * inv * sc[i] + bi[i];
    out[(size_t)row * D_ + i] = (__bf16)v;
  }
}

// ---------------------------------------------------------------------------
// W[K,N] fp32 -> Wt[N,K] bf16 (so GEMM B-fragments are contiguous in K).
// ---------------------------------------------------------------------------
__global__ __launch_bounds__(256) void k_transpose_bf16(
    const float* __restrict__ W, __bf16* __restrict__ Wt, int K, int N)
{
  __shared__ float tile[32][33];
  const int kb = blockIdx.y * 32, nb = blockIdx.x * 32;
  const int tx = threadIdx.x, ty = threadIdx.y;
  for (int r = ty; r < 32; r += 8)
    tile[r][tx] = W[(size_t)(kb + r) * N + nb + tx];
  __syncthreads();
  for (int r = ty; r < 32; r += 8)
    Wt[(size_t)(nb + r) * K + kb + tx] = (__bf16)tile[tx][r];
}

// ---------------------------------------------------------------------------
// WMMA GEMM: out[M,N] = A[M,K](bf16) @ Wt[N,K](bf16)^T + bias, with epilogue:
//  MODE 0: bf16 out            MODE 1: bf16 out + exact GELU
//  MODE 2: f32 out + residual  MODE 3: f32 out
// Block = 128x128 tile, 8 waves (2M x 4N), wave = 64x32 (4x2 frags), K step 32.
// Double-buffered LDS tiles filled with global_load_async_to_lds_b128; the
// async copy of tile k+1 overlaps the 8 WMMAs of tile k.
// ---------------------------------------------------------------------------
template <int MODE>
__global__ __launch_bounds__(256) void k_gemm(
    const __bf16* __restrict__ A, const __bf16* __restrict__ Wt,
    const float* __restrict__ bias, const float* __restrict__ resid,
    void* __restrict__ outv, int M, int N, int K)
{
  __shared__ __bf16 As[2][128 * 32];
  __shared__ __bf16 Bs[2][128 * 32];
  const int tid  = threadIdx.x;
  const int wave = tid >> 5, lane = tid & 31;
  const int wm = wave >> 2, wn = wave & 3;      // 2 x 4 waves
  const int lm = lane & 15, hi = lane >> 4;     // WMMA lane decomposition
  const int m0 = blockIdx.y * 128, n0 = blockIdx.x * 128;
  const int crow = tid >> 1, ccol = (tid & 1) * 16;   // cooperative copy

  // Per-thread global source rows and LDS destinations (32B segment each).
  const __bf16* gA = &A[(size_t)(m0 + crow) * K + ccol];
  const __bf16* gB = &Wt[(size_t)(n0 + crow) * K + ccol];
  uint32_t ldsA[2], ldsB[2];
  ldsA[0] = (uint32_t)(uintptr_t)&As[0][crow * 32 + ccol];
  ldsA[1] = (uint32_t)(uintptr_t)&As[1][crow * 32 + ccol];
  ldsB[0] = (uint32_t)(uintptr_t)&Bs[0][crow * 32 + ccol];
  ldsB[1] = (uint32_t)(uintptr_t)&Bs[1][crow * 32 + ccol];

  v8f c[4][2];
  #pragma unroll
  for (int i = 0; i < 4; ++i)
    #pragma unroll
    for (int j = 0; j < 2; ++j) { v8f z = {}; c[i][j] = z; }

  // Prologue: stage tile 0 into buffer 0.
  async_copy32(ldsA[0], gA);
  async_copy32(ldsB[0], gB);
  asm volatile("s_wait_asynccnt 0" ::: "memory");
  __syncthreads();

  for (int k0 = 0; k0 < K; k0 += 32) {
    const int cur = (k0 >> 5) & 1;
    // Issue async staging of the next K tile into the other buffer.
    if (k0 + 32 < K) {
      async_copy32(ldsA[cur ^ 1], gA + k0 + 32);
      async_copy32(ldsB[cur ^ 1], gB + k0 + 32);
    }

    v16bf a[4], bb[2];
    #pragma unroll
    for (int i = 0; i < 4; ++i) {
      // A-frag: lane row = lm, K halves at hi*8 and hi*8+16 (ISA 7.12.2)
      const __bf16* rp = &As[cur][(wm * 64 + i * 16 + lm) * 32];
      a[i] = frag_from2(rp + hi * 8, rp + hi * 8 + 16);
    }
    #pragma unroll
    for (int j = 0; j < 2; ++j) {
      // B-frag: lane col = lm, K halves at hi*8 and hi*8+16
      const __bf16* rp = &Bs[cur][(wn * 32 + j * 16 + lm) * 32];
      bb[j] = frag_from2(rp + hi * 8, rp + hi * 8 + 16);
    }
    #pragma unroll
    for (int i = 0; i < 4; ++i)
      #pragma unroll
      for (int j = 0; j < 2; ++j)
        c[i][j] = __builtin_amdgcn_wmma_f32_16x16x32_bf16(
            false, a[i], false, bb[j], (short)0, c[i][j], false, false);

    // Next buffer must be fully written (all waves) before the next K step;
    // current buffer's DS reads complete before the barrier as well.
    asm volatile("s_wait_asynccnt 0" ::: "memory");
    __syncthreads();
  }

  // Epilogue: C/D layout -> lane owns col lm, rows hi*8 + r
  #pragma unroll
  for (int i = 0; i < 4; ++i) {
    const int gm = m0 + wm * 64 + i * 16 + hi * 8;
    #pragma unroll
    for (int j = 0; j < 2; ++j) {
      const int gn = n0 + wn * 32 + j * 16 + lm;
      const float bvl = bias ? bias[gn] : 0.f;
      #pragma unroll
      for (int r = 0; r < 8; ++r) {
        float v = c[i][j][r] + bvl;
        const size_t idx = (size_t)(gm + r) * N + gn;
        if constexpr (MODE == 0) {
          ((__bf16*)outv)[idx] = (__bf16)v;
        } else if constexpr (MODE == 1) {
          float g = 0.5f * v * (1.0f + erff(v * 0.70710678118654752f));
          ((__bf16*)outv)[idx] = (__bf16)g;
        } else if constexpr (MODE == 2) {
          ((float*)outv)[idx] = resid[idx] + v;
        } else {
          ((float*)outv)[idx] = v;
        }
      }
    }
  }
}

// ---------------------------------------------------------------------------
// Flash attention, causal. One wave (32 threads) per (b, head, 16-row q tile).
// QK^T and PV via WMMA bf16; online softmax with per-lane row stats.
// ---------------------------------------------------------------------------
__global__ __launch_bounds__(32) void k_attn(
    const __bf16* __restrict__ Q, const __bf16* __restrict__ Kx,
    const __bf16* __restrict__ V, __bf16* __restrict__ Y)
{
  __shared__ __bf16 pS[16 * 32];   // P tile (16 rows x 32 keys)
  __shared__ __bf16 vt[64 * 32];   // V tile transposed: [d][key]
  const int lane = threadIdx.x;
  const int lm = lane & 15, hi = lane >> 4;
  const int q0 = blockIdx.x * 16;
  const int hd = blockIdx.y;
  const int b  = blockIdx.z;

  // Q fragments (A-matrix): row = q0+lm, two K chunks of 32 over DH=64
  const __bf16* qrow = Q + (size_t)(b * T_ + q0 + lm) * D_ + hd * DH_;
  v16bf qa[2];
  #pragma unroll
  for (int cch = 0; cch < 2; ++cch)
    qa[cch] = frag_from2(qrow + cch * 32 + hi * 8, qrow + cch * 32 + hi * 8 + 16);

  float mrow[8], lrow[8];
  v8f o[4];
  #pragma unroll
  for (int r = 0; r < 8; ++r) { mrow[r] = -3.0e38f; lrow[r] = 0.f; }
  #pragma unroll
  for (int t = 0; t < 4; ++t) { v8f z = {}; o[t] = z; }

  for (int kb0 = 0; kb0 < q0 + 16; kb0 += 32) {
    // ---- scores: two 16x16 tiles (keys kb0..kb0+15, kb0+16..kb0+31) ----
    v8f s[2];
    #pragma unroll
    for (int j = 0; j < 2; ++j) {
      const int kk = kb0 + j * 16 + lm;                 // key row for this lane
      const __bf16* krow = Kx + (size_t)(b * T_ + kk) * D_ + hd * DH_;
      v8f acc = {};
      #pragma unroll
      for (int cch = 0; cch < 2; ++cch) {
        // B-frag: col = key (lm), d halves at hi*8 and hi*8+16 within the chunk
        v16bf kf = frag_from2(krow + cch * 32 + hi * 8,
                              krow + cch * 32 + hi * 8 + 16);
        acc = __builtin_amdgcn_wmma_f32_16x16x32_bf16(
            false, qa[cch], false, kf, (short)0, acc, false, false);
      }
      s[j] = acc;
    }

    // ---- scale, causal mask, online softmax ----
    float pv0[8], pv1[8];
    #pragma unroll
    for (int r = 0; r < 8; ++r) {
      const int qr = q0 + hi * 8 + r;
      float s0 = s[0][r] * 0.125f;                       // 1/sqrt(64)
      float s1 = s[1][r] * 0.125f;
      if (kb0 + lm      > qr) s0 = -3.0e38f;
      if (kb0 + 16 + lm > qr) s1 = -3.0e38f;
      float mx = fmaxf(s0, s1);
      #pragma unroll
      for (int off = 1; off < 16; off <<= 1) mx = fmaxf(mx, __shfl_xor(mx, off, 16));
      const float mn = fmaxf(mrow[r], mx);
      const float p0 = __expf(s0 - mn), p1 = __expf(s1 - mn);
      pv0[r] = p0; pv1[r] = p1;
      float ps = p0 + p1;
      #pragma unroll
      for (int off = 1; off < 16; off <<= 1) ps += __shfl_xor(ps, off, 16);
      const float alpha = __expf(mrow[r] - mn);
      lrow[r] = lrow[r] * alpha + ps;
      mrow[r] = mn;
      #pragma unroll
      for (int t = 0; t < 4; ++t) o[t][r] *= alpha;
    }

    // ---- write P tile to LDS (row = q row, col = key) ----
    #pragma unroll
    for (int r = 0; r < 8; ++r) {
      pS[(hi * 8 + r) * 32 + lm]      = (__bf16)pv0[r];
      pS[(hi * 8 + r) * 32 + 16 + lm] = (__bf16)pv1[r];
    }
    // ---- stage V transposed: vt[d][key_local], lane owns one key row ----
    {
      const __bf16* vr = V + (size_t)(b * T_ + kb0 + lane) * D_ + hd * DH_;
      #pragma unroll 8
      for (int d = 0; d < 64; ++d) vt[d * 32 + lane] = vr[d];
    }
    asm volatile("s_wait_dscnt 0" ::: "memory");   // same-wave DS in-order; fence compiler
    __builtin_amdgcn_wave_barrier();

    // ---- PV: A = P (K=32 keys), B = V^T per 16-wide d tile ----
    v16bf pf = frag_from2(&pS[lm * 32 + hi * 8], &pS[lm * 32 + hi * 8 + 16]);
    #pragma unroll
    for (int t = 0; t < 4; ++t) {
      const __bf16* vp = &vt[(t * 16 + lm) * 32 + hi * 8];
      v16bf vf = frag_from2(vp, vp + 16);
      o[t] = __builtin_amdgcn_wmma_f32_16x16x32_bf16(
          false, pf, false, vf, (short)0, o[t], false, false);
    }
    asm volatile("s_wait_dscnt 0" ::: "memory");
    __builtin_amdgcn_wave_barrier();
  }

  // ---- normalize and write y (bf16, feeds proj GEMM) ----
  #pragma unroll
  for (int t = 0; t < 4; ++t)
    #pragma unroll
    for (int r = 0; r < 8; ++r) {
      const float v = o[t][r] / lrow[r];
      const size_t idx =
          (size_t)(b * T_ + q0 + hi * 8 + r) * D_ + hd * DH_ + t * 16 + lm;
      Y[idx] = (__bf16)v;
    }
}

// ---------------------------------------------------------------------------
// Host orchestration
// ---------------------------------------------------------------------------
extern "C" void kernel_launch(void* const* d_in, const int* in_sizes, int n_in,
                              void* d_out, int out_size, void* d_ws, size_t ws_size,
                              hipStream_t stream)
{
  (void)in_sizes; (void)n_in; (void)out_size; (void)ws_size;

  const float* lcd  = (const float*)d_in[0];
  const float* act  = (const float*)d_in[1];
  const float* pos  = (const float*)d_in[2];
  const float* Wemb = (const float*)d_in[3];
  const float* Wact = (const float*)d_in[4];
  const float* ln1s = (const float*)d_in[5];
  const float* ln1b = (const float*)d_in[6];
  const float* Wq   = (const float*)d_in[7];
  const float* bq   = (const float*)d_in[8];
  const float* Wk   = (const float*)d_in[9];
  const float* bk   = (const float*)d_in[10];
  const float* Wv   = (const float*)d_in[11];
  const float* bv   = (const float*)d_in[12];
  const float* Wp   = (const float*)d_in[13];
  const float* bp   = (const float*)d_in[14];
  const float* ln2s = (const float*)d_in[15];
  const float* ln2b = (const float*)d_in[16];
  const float* W1   = (const float*)d_in[17];
  const float* b1   = (const float*)d_in[18];
  const float* W2   = (const float*)d_in[19];
  const float* b2   = (const float*)d_in[20];
  const float* lnfs = (const float*)d_in[21];
  const float* lnfb = (const float*)d_in[22];
  const float* Wh   = (const float*)d_in[23];
  const float* bh   = (const float*)d_in[24];

  // Workspace layout (~96 MB total)
  char* ws = (char*)d_ws;
  float*  h     = (float*)(ws);                              // 16 MB f32 residual stream
  __bf16* zbf   = (__bf16*)(ws + ((size_t)16 << 20));        //  8 MB LN output
  __bf16* qbf   = (__bf16*)(ws + ((size_t)24 << 20));        //  8 MB
  __bf16* kbf   = (__bf16*)(ws + ((size_t)32 << 20));        //  8 MB
  __bf16* vbf   = (__bf16*)(ws + ((size_t)40 << 20));        //  8 MB
  __bf16* ybf   = (__bf16*)(ws + ((size_t)48 << 20));        //  8 MB attn out
  __bf16* hidbf = (__bf16*)(ws + ((size_t)56 << 20));        // 32 MB MLP hidden
  __bf16* wt    = (__bf16*)(ws + ((size_t)88 << 20));        //  8 MB bf16 W^T slab

  const dim3 tb(32, 8);
  const dim3 gD  (D_  / 128, M_ / 128);   // N=1024 GEMMs
  const dim3 gFF (FF_ / 128, M_ / 128);   // N=4096 GEMM
  const dim3 gE  (E_  / 128, M_ / 128);   // head GEMM

  k_embed<<<M_, 256, 0, stream>>>(lcd, act, pos, Wemb, Wact, h);

  for (int l = 0; l < NL_; ++l) {
    const size_t oDD = (size_t)l * D_ * D_;
    const size_t oDF = (size_t)l * D_ * FF_;

    k_ln<<<M_, 256, 0, stream>>>(h, ln1s + (size_t)l * D_, ln1b + (size_t)l * D_, zbf);

    k_transpose_bf16<<<dim3(D_/32, D_/32), tb, 0, stream>>>(Wq + oDD, wt, D_, D_);
    k_gemm<0><<<gD, 256, 0, stream>>>(zbf, wt, bq + (size_t)l * D_, nullptr, qbf, M_, D_, D_);

    k_transpose_bf16<<<dim3(D_/32, D_/32), tb, 0, stream>>>(Wk + oDD, wt, D_, D_);
    k_gemm<0><<<gD, 256, 0, stream>>>(zbf, wt, bk + (size_t)l * D_, nullptr, kbf, M_, D_, D_);

    k_transpose_bf16<<<dim3(D_/32, D_/32), tb, 0, stream>>>(Wv + oDD, wt, D_, D_);
    k_gemm<0><<<gD, 256, 0, stream>>>(zbf, wt, bv + (size_t)l * D_, nullptr, vbf, M_, D_, D_);

    k_attn<<<dim3(T_/16, NH_, B_), 32, 0, stream>>>(qbf, kbf, vbf, ybf);

    k_transpose_bf16<<<dim3(D_/32, D_/32), tb, 0, stream>>>(Wp + oDD, wt, D_, D_);
    k_gemm<2><<<gD, 256, 0, stream>>>(ybf, wt, bp + (size_t)l * D_, h, h, M_, D_, D_);

    k_ln<<<M_, 256, 0, stream>>>(h, ln2s + (size_t)l * D_, ln2b + (size_t)l * D_, zbf);

    k_transpose_bf16<<<dim3(FF_/32, D_/32), tb, 0, stream>>>(W1 + oDF, wt, D_, FF_);
    k_gemm<1><<<gFF, 256, 0, stream>>>(zbf, wt, b1 + (size_t)l * FF_, nullptr, hidbf, M_, FF_, D_);

    k_transpose_bf16<<<dim3(D_/32, FF_/32), tb, 0, stream>>>(W2 + oDF, wt, FF_, D_);
    k_gemm<2><<<gD, 256, 0, stream>>>(hidbf, wt, b2 + (size_t)l * D_, h, h, M_, D_, FF_);
  }

  k_ln<<<M_, 256, 0, stream>>>(h, lnfs, lnfb, zbf);
  k_transpose_bf16<<<dim3(E_/32, D_/32), tb, 0, stream>>>(Wh, wt, D_, E_);
  k_gemm<3><<<gE, 256, 0, stream>>>(zbf, wt, bh, nullptr, d_out, M_, E_, D_);
}
